// PhysSelfAttention_85040352461004
// MI455X (gfx1250) — matrix operand order repaired and checked by reading, not compiled, not run on hardware
//
#include <hip/hip_runtime.h>
#include <hip/hip_bf16.h>

// ---------------------------------------------------------------------------
// PhysSelfAttention for MI455X (gfx1250): bf16 WMMA GEMMs + flash attention.
//   B=2, P=2048, C=768, H=12, hd=64
// GEMMs: 32Mx64N per wave, software-pipelined fragment loads (loads of K-step
// k+1 overlap the 8 WMMAs of step k). __launch_bounds__(256) so the allocator
// keeps accumulators + double-buffered fragments resident (no scratch spill).
// Attention: register-buffered K/V staging pipeline over 32-key blocks,
// online softmax, never materializes att.
// ---------------------------------------------------------------------------

#define BATCH 2
#define SEQ   2048
#define EMBD  768
#define HEADS 12
#define HD    64
#define QKVN  (3 * EMBD)    // 2304
#define ROWS  (BATCH * SEQ) // 4096

typedef __bf16 bf16_t;
typedef __attribute__((ext_vector_type(16))) __bf16 v16bf;
typedef __attribute__((ext_vector_type(8)))  __bf16 v8bf;
typedef __attribute__((ext_vector_type(8)))  float  v8f;

__device__ __forceinline__ v8bf load8g(const bf16_t* p) {
    return *reinterpret_cast<const v8bf*>(p);
}
__device__ __forceinline__ v16bf mk16(v8bf lo, v8bf hi) {
    return __builtin_shufflevector(lo, hi, 0,1,2,3,4,5,6,7,8,9,10,11,12,13,14,15);
}
__device__ __forceinline__ v8f wmma_bf16(v16bf a, v16bf b, v8f c) {
    // D = A(16x32) * B(32x16) + C(16x16), f32 accumulate
    return __builtin_amdgcn_wmma_f32_16x16x32_bf16(false, a, false, b, (short)0, c,
                                                   false, false);
}

// ------------------------- conversion kernels ------------------------------

__global__ __launch_bounds__(256) void k_cvt(const float* __restrict__ src,
                                             bf16_t* __restrict__ dst, int n) {
    int i = blockIdx.x * 256 + threadIdx.x;
    if (i < n) dst[i] = (bf16_t)src[i];
}

// src [K][N] fp32 row-major  ->  dst [N][K] bf16 (transposed, so WMMA B-frags
// become contiguous 16-byte loads along k)
__global__ __launch_bounds__(256) void k_cvt_t(const float* __restrict__ src,
                                               bf16_t* __restrict__ dst,
                                               int K, int N) {
    int i = blockIdx.x * 256 + threadIdx.x;
    if (i < K * N) {
        int k = i / N, n = i % N;
        dst[(size_t)n * K + k] = (bf16_t)src[i];
    }
}

// --------------------- pipelined 32Mx64N GEMM core -------------------------
// A [M][768] bf16 row-major, Bt [N][768] bf16 (weights pre-transposed).
// Wave computes 32 M-rows x 64 N-cols: 8 f32 accumulators.
// Next K-step fragments are fetched into fresh registers before the current
// step's WMMAs issue, so global-load latency overlaps the matrix pipe.
__device__ __forceinline__ void gemm_tile_acc(const bf16_t* __restrict__ A,
                                              const bf16_t* __restrict__ Bt,
                                              int Mbase, int Nbase,
                                              v8f acc[2][4]) {
    const int lane = threadIdx.x & 31;
    const int hi = lane >> 4, col = lane & 15;
    const int abase = hi * 8;
    const bf16_t* ar0 = A + (size_t)(Mbase + col) * EMBD;        // A-frag: M = lane&15
    const bf16_t* ar1 = A + (size_t)(Mbase + 16 + col) * EMBD;
    const bf16_t* brow[4];
#pragma unroll
    for (int nb = 0; nb < 4; ++nb)
        brow[nb] = Bt + (size_t)(Nbase + nb * 16 + col) * EMBD + hi * 8;

    // prologue: K-step 0 fragments
    v16bf a0 = mk16(load8g(ar0 + abase), load8g(ar0 + 16 + abase));
    v16bf a1 = mk16(load8g(ar1 + abase), load8g(ar1 + 16 + abase));
    v16bf b[4];
#pragma unroll
    for (int nb = 0; nb < 4; ++nb)
        b[nb] = mk16(load8g(brow[nb]), load8g(brow[nb] + 16));

    for (int k0 = 32; k0 < EMBD; k0 += 32) {
        // fetch next K-step into fresh registers (overlaps WMMAs below)
        v16bf a0n = mk16(load8g(ar0 + k0 + abase), load8g(ar0 + k0 + 16 + abase));
        v16bf a1n = mk16(load8g(ar1 + k0 + abase), load8g(ar1 + k0 + 16 + abase));
        v16bf bn[4];
#pragma unroll
        for (int nb = 0; nb < 4; ++nb)
            bn[nb] = mk16(load8g(brow[nb] + k0), load8g(brow[nb] + k0 + 16));
        // compute current K-step: 8 WMMAs, B fragments reused across 2 M-tiles
#pragma unroll
        for (int nb = 0; nb < 4; ++nb) acc[0][nb] = wmma_bf16(a0, b[nb], acc[0][nb]);
#pragma unroll
        for (int nb = 0; nb < 4; ++nb) acc[1][nb] = wmma_bf16(a1, b[nb], acc[1][nb]);
        a0 = a0n; a1 = a1n;
#pragma unroll
        for (int nb = 0; nb < 4; ++nb) b[nb] = bn[nb];
    }
    // epilogue K-step
#pragma unroll
    for (int nb = 0; nb < 4; ++nb) acc[0][nb] = wmma_bf16(a0, b[nb], acc[0][nb]);
#pragma unroll
    for (int nb = 0; nb < 4; ++nb) acc[1][nb] = wmma_bf16(a1, b[nb], acc[1][nb]);
}

// ------------------------------ QKV GEMM -----------------------------------
// out: q/k/v bf16 each laid out [B][H][P][64]
__global__ __launch_bounds__(256) void gemm_qkv(const bf16_t* __restrict__ A,
                                                const bf16_t* __restrict__ Bt,
                                                const float* __restrict__ bias,
                                                bf16_t* __restrict__ qb,
                                                bf16_t* __restrict__ kb,
                                                bf16_t* __restrict__ vb) {
    const int lane = threadIdx.x & 31;
    const int wave = threadIdx.x >> 5;
    const int hi   = lane >> 4, col = lane & 15;
    const int Mbase = blockIdx.x * 256 + wave * 32;
    const int Nbase = blockIdx.y * 64;

    v8f acc[2][4] = {};
    gemm_tile_acc(A, Bt, Mbase, Nbase, acc);

    // Epilogue: C-layout is (M = r + 8*hi, N = col); scatter into head-major q/k/v
#pragma unroll
    for (int mt = 0; mt < 2; ++mt) {
#pragma unroll
        for (int nb = 0; nb < 4; ++nb) {
#pragma unroll
            for (int r = 0; r < 8; ++r) {
                int m = Mbase + mt * 16 + r + 8 * hi;
                int n = Nbase + nb * 16 + col;
                float v = acc[mt][nb][r] + bias[n];
                int which = n / EMBD, c = n % EMBD;
                int h = c >> 6, d = c & 63;
                int b = m >> 11, p = m & 2047;
                bf16_t* dst = (which == 0) ? qb : (which == 1) ? kb : vb;
                dst[(((size_t)(b * HEADS + h) * SEQ) + p) * HD + d] = (bf16_t)v;
            }
        }
    }
}

// ---------------------------- projection GEMM ------------------------------
// A: yb [4096][768] bf16, Bt: W_proj^T [768][768] bf16, bias fp32[768]
// out: fp32 [4096][768]
__global__ __launch_bounds__(256) void gemm_proj(const bf16_t* __restrict__ A,
                                                 const bf16_t* __restrict__ Bt,
                                                 const float* __restrict__ bias,
                                                 float* __restrict__ out) {
    const int lane = threadIdx.x & 31;
    const int wave = threadIdx.x >> 5;
    const int hi = lane >> 4, col = lane & 15;
    const int Mbase = blockIdx.x * 256 + wave * 32;
    const int Nbase = blockIdx.y * 64;

    v8f acc[2][4] = {};
    gemm_tile_acc(A, Bt, Mbase, Nbase, acc);

#pragma unroll
    for (int mt = 0; mt < 2; ++mt) {
#pragma unroll
        for (int nb = 0; nb < 4; ++nb) {
#pragma unroll
            for (int r = 0; r < 8; ++r) {
                int m = Mbase + mt * 16 + r + 8 * hi;
                int n = Nbase + nb * 16 + col;
                out[(size_t)m * EMBD + n] = acc[mt][nb][r] + bias[n];
            }
        }
    }
}

// --------------------------- flash attention -------------------------------
// q/k/v: [B][H][P][64] bf16.  gamma: [P][P] fp32.  yb out: [B][P][768] bf16.
// WG = 256 threads (8 waves); each wave owns 16 q rows; 32-key blocks with
// register-buffered global->LDS staging pipeline.
__global__ __launch_bounds__(256) void attn_kernel(const bf16_t* __restrict__ qb,
                                                   const bf16_t* __restrict__ kb,
                                                   const bf16_t* __restrict__ vb,
                                                   const float* __restrict__ gamma,
                                                   bf16_t* __restrict__ yb) {
    __shared__ __align__(16) bf16_t Ks[32 * 64];      // K block  [key][d]
    __shared__ __align__(16) bf16_t Vt[64 * 40];      // V block transposed [d][key], padded
    __shared__ __align__(16) bf16_t Pr[8][16 * 32];   // per-wave prob tile [q][key]

    const int lane = threadIdx.x & 31;
    const int wave = threadIdx.x >> 5;
    const int hi = lane >> 4, col = lane & 15;
    const int bh = blockIdx.y;                 // 0..23 = b*12+h
    const int b  = bh / HEADS, h = bh % HEADS;
    const int qt = blockIdx.x * 128 + wave * 16;

    const size_t head_off = (size_t)bh * SEQ * HD;

    // staging assignment: each thread owns 8 contiguous bf16 of one K row + V row
    const int srow = threadIdx.x >> 3;
    const int scp  = (threadIdx.x & 7) * 8;
    const bf16_t* ksrc = kb + head_off + (size_t)srow * HD + scp;
    const bf16_t* vsrc = vb + head_off + (size_t)srow * HD + scp;

    // Q fragments (A-layout), head dim 64 = two K=32 chunks
    const bf16_t* qr = qb + head_off + (size_t)(qt + col) * HD;
    v16bf aq0 = mk16(load8g(qr + hi * 16),      load8g(qr + 8 + hi * 16));
    v16bf aq1 = mk16(load8g(qr + 32 + hi * 16), load8g(qr + 40 + hi * 16));

    float mrow[8], lrow[8];
    v8f yacc[4] = {};
#pragma unroll
    for (int r = 0; r < 8; ++r) { mrow[r] = -3.0e38f; lrow[r] = 0.0f; }

    const float scale = 0.125f; // 1/sqrt(64)

    // prologue: fetch key-block 0 into registers
    v8bf kreg = load8g(ksrc);
    v8bf vreg = load8g(vsrc);

    for (int key0 = 0; key0 < SEQ; key0 += 32) {
        // ---- commit staged registers to LDS (K plain, V transposed) ----
        *reinterpret_cast<v8bf*>(&Ks[srow * 64 + scp]) = kreg;
#pragma unroll
        for (int j = 0; j < 8; ++j) Vt[(scp + j) * 40 + srow] = vreg[j];
        __syncthreads();

        // ---- issue next block's global loads; they fly during compute ----
        if (key0 + 32 < SEQ) {
            kreg = load8g(ksrc + (size_t)(key0 + 32) * HD);
            vreg = load8g(vsrc + (size_t)(key0 + 32) * HD);
            // next gamma tile for this wave's rows -> L2/L0 warm
            __builtin_prefetch(gamma + (size_t)(qt + col) * SEQ + key0 + 32 + hi * 16, 0, 1);
        }

        // ---- scores: S(16q x 32k) as two 16x16 WMMA accumulators ----
        v8f s[2];
#pragma unroll
        for (int kt = 0; kt < 2; ++kt) {
            const bf16_t* kp = &Ks[(kt * 16 + col) * 64 + hi * 16]; // B-frag: n=key, k=d
            v16bf b0 = mk16(load8g(kp),      load8g(kp + 8));       // d in [0,32)
            v16bf b1 = mk16(load8g(kp + 32), load8g(kp + 40));      // d in [32,64)
            v8f c = {};
            c = wmma_bf16(aq0, b0, c);
            c = wmma_bf16(aq1, b1, c);
            s[kt] = c;
        }

        // ---- online softmax (per q row = (r, half); 16-lane reductions) ----
        float alpha[8];
#pragma unroll
        for (int r = 0; r < 8; ++r) {
            int qrow = qt + r + 8 * hi;
            const float* g = gamma + (size_t)qrow * SEQ + key0;
            float v0 = s[0][r] * scale + g[col];
            float v1 = s[1][r] * scale + g[16 + col];
            v0 = fminf(fmaxf(v0, -1e9f), 1e9f);
            v1 = fminf(fmaxf(v1, -1e9f), 1e9f);
            float bm = fmaxf(v0, v1);
            bm = fmaxf(bm, __shfl_xor(bm, 1, 32));
            bm = fmaxf(bm, __shfl_xor(bm, 2, 32));
            bm = fmaxf(bm, __shfl_xor(bm, 4, 32));
            bm = fmaxf(bm, __shfl_xor(bm, 8, 32));
            float mn = fmaxf(mrow[r], bm);
            float al = __expf(mrow[r] - mn);
            float p0 = __expf(v0 - mn);
            float p1 = __expf(v1 - mn);
            float ps = p0 + p1;
            ps += __shfl_xor(ps, 1, 32);
            ps += __shfl_xor(ps, 2, 32);
            ps += __shfl_xor(ps, 4, 32);
            ps += __shfl_xor(ps, 8, 32);
            lrow[r] = lrow[r] * al + ps;
            mrow[r] = mn;
            alpha[r] = al;
            // C-layout -> LDS [q][key] so we can reload in A-layout
            Pr[wave][(r + 8 * hi) * 32 + col]      = (bf16_t)p0;
            Pr[wave][(r + 8 * hi) * 32 + 16 + col] = (bf16_t)p1;
        }
#pragma unroll
        for (int dt = 0; dt < 4; ++dt)
#pragma unroll
            for (int r = 0; r < 8; ++r) yacc[dt][r] *= alpha[r];

        // ---- P fragment (A-layout 16x32) from per-wave LDS tile ----
        const bf16_t* pp = &Pr[wave][col * 32 + hi * 16];
        v16bf pa = mk16(load8g(pp), load8g(pp + 8));

        // ---- PV: y(16q x 64d) += P(16x32) x V(32x16) per 16-wide d chunk ----
#pragma unroll
        for (int dt = 0; dt < 4; ++dt) {
            const bf16_t* vp = &Vt[(dt * 16 + col) * 40 + hi * 16]; // B-frag: n=d, k=key
            v16bf vfr = mk16(load8g(vp), load8g(vp + 8));
            yacc[dt] = wmma_bf16(pa, vfr, yacc[dt]);
        }
        __syncthreads();
    }

    // ---- normalize and store back to [B][P][C] bf16 for the projection ----
#pragma unroll
    for (int r = 0; r < 8; ++r) {
        float inv = 1.0f / lrow[r];
        int qrow = qt + r + 8 * hi;
        size_t base = ((size_t)(b * SEQ + qrow)) * EMBD + h * HD;
#pragma unroll
        for (int dt = 0; dt < 4; ++dt)
            yb[base + dt * 16 + col] = (bf16_t)(yacc[dt][r] * inv);
    }
}

// ------------------------------- launcher ----------------------------------
extern "C" void kernel_launch(void* const* d_in, const int* in_sizes, int n_in,
                              void* d_out, int out_size, void* d_ws, size_t ws_size,
                              hipStream_t stream) {
    (void)in_sizes; (void)n_in; (void)out_size; (void)ws_size;
    const float* x       = (const float*)d_in[0];
    const float* gamma   = (const float*)d_in[1];
    const float* W_attn  = (const float*)d_in[2];
    const float* b_attn  = (const float*)d_in[3];
    const float* W_proj  = (const float*)d_in[4];
    const float* b_proj  = (const float*)d_in[5];

    // workspace carve-up (all sizes multiples of 256B; total ~36.1 MB)
    char* w = (char*)d_ws;
    bf16_t* xb  = (bf16_t*)w; w += (size_t)ROWS * EMBD * 2;   // x in bf16
    bf16_t* Wat = (bf16_t*)w; w += (size_t)QKVN * EMBD * 2;   // W_attn^T bf16
    bf16_t* Wpt = (bf16_t*)w; w += (size_t)EMBD * EMBD * 2;   // W_proj^T bf16
    bf16_t* qb  = (bf16_t*)w; w += (size_t)ROWS * EMBD * 2;   // q [B,H,P,64]
    bf16_t* kb  = (bf16_t*)w; w += (size_t)ROWS * EMBD * 2;   // k [B,H,P,64]
    bf16_t* vb  = (bf16_t*)w; w += (size_t)ROWS * EMBD * 2;   // v [B,H,P,64]
    bf16_t* yb  = (bf16_t*)w; w += (size_t)ROWS * EMBD * 2;   // attn out [B,P,C]

    k_cvt  <<<(ROWS * EMBD) / 256, 256, 0, stream>>>(x, xb, ROWS * EMBD);
    k_cvt_t<<<(EMBD * QKVN) / 256, 256, 0, stream>>>(W_attn, Wat, EMBD, QKVN);
    k_cvt_t<<<(EMBD * EMBD) / 256, 256, 0, stream>>>(W_proj, Wpt, EMBD, EMBD);

    gemm_qkv<<<dim3(ROWS / 256, QKVN / 64), 256, 0, stream>>>(xb, Wat, b_attn,
                                                              qb, kb, vb);

    attn_kernel<<<dim3(SEQ / 128, BATCH * HEADS), 256, 0, stream>>>(qb, kb, vb,
                                                                    gamma, yb);

    gemm_proj<<<dim3(ROWS / 256, EMBD / 64), 256, 0, stream>>>(yb, Wpt, b_proj,
                                                               (float*)d_out);
}
